// multiHeadAttentionBlock_59674275610707
// MI455X (gfx1250) — compile-verified, matrix operand order checked by reading, not run on hardware
//
#include <hip/hip_runtime.h>
#include <hip/hip_bf16.h>

// Problem constants (match reference)
#define B_  4
#define S_  2048
#define E_  1024
#define H_  16
#define HD_ 64

typedef _Float16 v16h __attribute__((ext_vector_type(16)));
typedef _Float16 v8h  __attribute__((ext_vector_type(8)));
typedef _Float16 v4h  __attribute__((ext_vector_type(4)));
typedef float    v8f  __attribute__((ext_vector_type(8)));
typedef float    v4f  __attribute__((ext_vector_type(4)));

// ---------------------------------------------------------------------------
// CDNA5 async global->LDS copy (ASYNCcnt-tracked), with sync fallback.
// Builtin signature (from compiler diagnostic): pointee is a 16-byte GCC-style
// int vector; src in addrspace(1), dst in addrspace(3).
// ---------------------------------------------------------------------------
#if defined(__has_builtin)
#if __has_builtin(__builtin_amdgcn_global_load_async_to_lds_b128) && \
    __has_builtin(__builtin_amdgcn_s_wait_asynccnt)
#define HAVE_ASYNC_LDS 1
#endif
#endif
#ifndef HAVE_ASYNC_LDS
#define HAVE_ASYNC_LDS 0
#endif

typedef int gv4i __attribute__((vector_size(16)));
typedef __attribute__((address_space(1))) gv4i* gv4i_gptr;
typedef __attribute__((address_space(3))) gv4i* gv4i_lptr;

__device__ __forceinline__ void copy16_g2lds(_Float16* dst,
                                             const _Float16* src) {
#if HAVE_ASYNC_LDS
  __builtin_amdgcn_global_load_async_to_lds_b128((gv4i_gptr)src, (gv4i_lptr)dst,
                                                 0, 0);
#else
  *(v8h*)dst = *(const v8h*)src;
#endif
}

__device__ __forceinline__ void wait_async_lds() {
#if HAVE_ASYNC_LDS
  __builtin_amdgcn_s_wait_asynccnt(0);
#endif
}

__device__ __forceinline__ v8f zero8f() {
  v8f z;
#pragma unroll
  for (int i = 0; i < 8; ++i) z[i] = 0.0f;
  return z;
}

__device__ __forceinline__ v8f wmma_f16f32(v16h a, v16h b, v8f c) {
  // 8 args: (neg_a, A, neg_b, B, c_mod, C, reuse_a, reuse_b)
  return __builtin_amdgcn_wmma_f32_16x16x32_f16(false, a, false, b, (short)0, c,
                                                false, false);
}

// Load a 16x32 (f16) A/B fragment from LDS rows laid out [row][k] with row
// stride `stride_h` halves. Matches the CDNA5 16-bit A-matrix layout:
//   lanes 0-15  : row = lane,    halves 0-7 = K 0-7,  halves 8-15 = K 16-23
//   lanes 16-31 : row = lane-16, halves 0-7 = K 8-15, halves 8-15 = K 24-31
__device__ __forceinline__ v16h load_frag_lds(const _Float16* base, int stride_h,
                                              int lane) {
  const int r    = lane & 15;
  const int ksel = (lane >> 4) * 8;
  v8h lo = *(const v8h*)(base + r * stride_h + ksel);
  v8h hi = *(const v8h*)(base + r * stride_h + 16 + ksel);
  v16h out;
#pragma unroll
  for (int i = 0; i < 8; ++i) { out[i] = lo[i]; out[8 + i] = hi[i]; }
  return out;
}

// ---------------------------------------------------------------------------
// fp32 -> f16 conversion (weights)
// ---------------------------------------------------------------------------
__global__ void cvt_f32_f16_kernel(const float* __restrict__ src,
                                   _Float16* __restrict__ dst, int n4) {
  int i = blockIdx.x * blockDim.x + threadIdx.x;
  if (i < n4) {
    v4f x = ((const v4f*)src)[i];
    v4h y;
#pragma unroll
    for (int e = 0; e < 4; ++e) y[e] = (_Float16)x[e];
    ((v4h*)dst)[i] = y;
  }
}

// ---------------------------------------------------------------------------
// GEMM: Y[M,E] = X[M,E] @ W[E,E]^T + b   (nn.Linear semantics)
//   B-matrix element (k, n) = W[n, k] -> column n of B is contiguous row n of
//   W, so B fragments stage exactly like A fragments.
// Block: 128 threads = 4 waves (2x2), tile 128x128, K slice 32.
// Double-buffered LDS; f16 tiles staged with async global->LDS; the fp32
// activation tile is software-pipelined through VGPRs (load slice k+1 before
// the WMMA block, convert+ds_store after it).
// ---------------------------------------------------------------------------
template <typename AT, bool OUT_F16>
__global__ __launch_bounds__(128) void gemm_xwT_kernel(
    const AT* __restrict__ A, const _Float16* __restrict__ W,
    const float* __restrict__ bias, void* __restrict__ Yv) {
  constexpr int KC = 32;
  constexpr int LDS_S = KC + 8;  // 40 halves: pad to spread LDS banks
  constexpr int NK = E_ / KC;
  __shared__ __align__(16) _Float16 As[2][128 * LDS_S];
  __shared__ __align__(16) _Float16 Bs[2][128 * LDS_S];

  const int tid  = threadIdx.x;
  const int lane = tid & 31;
  const int wave = tid >> 5;
  const int wr   = (wave >> 1) * 64;  // wave row offset in tile
  const int wc   = (wave & 1) * 64;   // wave col offset in tile
  const int m0   = blockIdx.x * 128;
  const int n0   = blockIdx.y * 128;

  v8f acc[4][4];
#pragma unroll
  for (int i = 0; i < 4; ++i)
#pragma unroll
    for (int j = 0; j < 4; ++j) acc[i][j] = zero8f();

  v4f areg[8];  // in-flight fp32 A slice (pipelined)

  auto stageB = [&](int buf, int k0) {
#pragma unroll
    for (int i = 0; i < 4; ++i) {
      int f = i * 128 + tid, row = f >> 2, col = (f & 3) * 8;
      copy16_g2lds(&Bs[buf][row * LDS_S + col],
                   W + (size_t)(n0 + row) * E_ + k0 + col);
    }
  };
  auto stageA_f16 = [&](int buf, int k0) {
    if constexpr (sizeof(AT) == 2) {
#pragma unroll
      for (int i = 0; i < 4; ++i) {
        int f = i * 128 + tid, row = f >> 2, col = (f & 3) * 8;
        copy16_g2lds(&As[buf][row * LDS_S + col],
                     (const _Float16*)A + (size_t)(m0 + row) * E_ + k0 + col);
      }
    }
  };
  auto loadA_f32 = [&](int k0) {  // issue global loads only
    if constexpr (sizeof(AT) == 4) {
#pragma unroll
      for (int i = 0; i < 8; ++i) {
        int f = i * 128 + tid, row = f >> 3, col = (f & 7) * 4;
        areg[i] =
            *(const v4f*)((const float*)A + (size_t)(m0 + row) * E_ + k0 + col);
      }
    }
  };
  auto storeA_f32 = [&](int buf) {  // convert + LDS store
    if constexpr (sizeof(AT) == 4) {
#pragma unroll
      for (int i = 0; i < 8; ++i) {
        int f = i * 128 + tid, row = f >> 3, col = (f & 7) * 4;
        v4h y;
#pragma unroll
        for (int e = 0; e < 4; ++e) y[e] = (_Float16)areg[i][e];
        *(v4h*)&As[buf][row * LDS_S + col] = y;
      }
    }
  };

  // prologue: slice 0 -> buffer 0
  if constexpr (sizeof(AT) == 4) {
    loadA_f32(0);
    storeA_f32(0);
  } else {
    stageA_f16(0, 0);
  }
  stageB(0, 0);

  for (int kk = 0; kk < NK; ++kk) {
    const int buf = kk & 1;
    wait_async_lds();  // async tiles for `buf` complete
    __syncthreads();   // + compiler-inserted dscnt wait for ds_store path
    const bool more = (kk + 1) < NK;
    if (more) {
      stageA_f16(buf ^ 1, (kk + 1) * KC);  // async issue, no wait
      stageB(buf ^ 1, (kk + 1) * KC);      // async issue, no wait
      loadA_f32((kk + 1) * KC);            // global->VGPR issue, wait deferred
    }

    v16h af[4], bf[4];
#pragma unroll
    for (int i = 0; i < 4; ++i)
      af[i] = load_frag_lds(&As[buf][(wr + i * 16) * LDS_S], LDS_S, lane);
#pragma unroll
    for (int j = 0; j < 4; ++j)
      bf[j] = load_frag_lds(&Bs[buf][(wc + j * 16) * LDS_S], LDS_S, lane);
#pragma unroll
    for (int i = 0; i < 4; ++i)
#pragma unroll
      for (int j = 0; j < 4; ++j)
        acc[i][j] = wmma_f16f32(af[i], bf[j], acc[i][j]);

    if (more) storeA_f32(buf ^ 1);  // loadcnt wait lands behind the WMMAs
  }

  // ---- epilogue: C layout M = r + 8*(lane>=16), N = lane & 15 ----
  const int hf = lane >> 4, nl = lane & 15;
#pragma unroll
  for (int j = 0; j < 4; ++j) {
    int n = n0 + wc + j * 16 + nl;
    float bv = bias[n];
#pragma unroll
    for (int i = 0; i < 4; ++i) {
      int mb = m0 + wr + i * 16 + hf * 8;
#pragma unroll
      for (int r = 0; r < 8; ++r) {
        float val = acc[i][j][r] + bv;
        size_t off = (size_t)(mb + r) * E_ + n;
        if constexpr (OUT_F16)
          ((_Float16*)Yv)[off] = (_Float16)val;
        else
          ((float*)Yv)[off] = val;
      }
    }
  }
}

// ---------------------------------------------------------------------------
// Flash attention, causal, per (b, h). Block = 128 queries (4 waves x 32),
// streaming 64-key blocks. WMMA f16->f32; softmax stats in f32.
// Head views: Qh[s, d] = Q16[(b*S + s)*E + h*64 + d]  (row stride E).
// ---------------------------------------------------------------------------
__global__ __launch_bounds__(128) void attn_flash_kernel(
    const _Float16* __restrict__ Q, const _Float16* __restrict__ K,
    const _Float16* __restrict__ V, _Float16* __restrict__ O) {
  __shared__ __align__(16) _Float16 Ks[64 * 72];     // [key][d]
  __shared__ __align__(16) _Float16 Vt[64 * 72];     // [d][key] (transposed)
  __shared__ __align__(16) _Float16 Ps[4][32 * 72];  // per-wave P [q][key]

  const int tid  = threadIdx.x;
  const int lane = tid & 31;
  const int wave = tid >> 5;
  const int bh = blockIdx.y, b = bh >> 4, h = bh & 15;
  const int q0 = blockIdx.x * 128;
  const int qw = q0 + wave * 32;  // this wave's first query
  const int hf = lane >> 4, nl = lane & 15, ksel = hf * 8;

  const _Float16* Qp = Q + (size_t)b * S_ * E_ + h * HD_;
  const _Float16* Kp = K + (size_t)b * S_ * E_ + h * HD_;
  const _Float16* Vp = V + (size_t)b * S_ * E_ + h * HD_;

  // Q A-fragments straight from global: rows are contiguous over d.
  v16h af[2][2];  // [q-tile][d-chunk of 32]
#pragma unroll
  for (int ti = 0; ti < 2; ++ti)
#pragma unroll
    for (int c = 0; c < 2; ++c) {
      const _Float16* base =
          Qp + (size_t)(qw + ti * 16 + nl) * E_ + c * 32 + ksel;
      v8h lo = *(const v8h*)base;
      v8h hi = *(const v8h*)(base + 16);
#pragma unroll
      for (int i = 0; i < 8; ++i) {
        af[ti][c][i] = lo[i];
        af[ti][c][8 + i] = hi[i];
      }
    }

  float m_run[2][8], l_run[2][8];
  v8f cacc[2][4];  // [q-tile][d-tile] context accumulators
#pragma unroll
  for (int ti = 0; ti < 2; ++ti) {
#pragma unroll
    for (int r = 0; r < 8; ++r) { m_run[ti][r] = -3.0e38f; l_run[ti][r] = 0.0f; }
#pragma unroll
    for (int j = 0; j < 4; ++j) cacc[ti][j] = zero8f();
  }

  const int nkb = (q0 >> 6) + 2;  // causal: key blocks covering [0, q0+127]
  for (int kb = 0; kb < nkb; ++kb) {
    const int j0 = kb * 64;
    __syncthreads();
    // ---- stage K tile [64 keys][64 d] via async global->LDS ----
#pragma unroll
    for (int i = 0; i < 4; ++i) {
      int f = i * 128 + tid, row = f >> 3, col = (f & 7) * 8;
      copy16_g2lds(&Ks[row * 72 + col], Kp + (size_t)(j0 + row) * E_ + col);
    }
    // ---- stage V transposed [64 d][64 keys] (needs VGPR transpose) ----
#pragma unroll
    for (int i = 0; i < 4; ++i) {
      int f = i * 128 + tid, row = f >> 3, col = (f & 7) * 8;
      v8h x = *(const v8h*)(Vp + (size_t)(j0 + row) * E_ + col);
#pragma unroll
      for (int e = 0; e < 8; ++e) Vt[(col + e) * 72 + row] = x[e];
    }
    wait_async_lds();
    __syncthreads();

    // prefetch next key block while this one is being consumed
    if (kb + 1 < nkb) {
      const size_t nrow = (size_t)(j0 + 64 + (tid >> 1)) * E_ + (tid & 1) * 32;
      __builtin_prefetch(Kp + nrow, 0, 3);
      __builtin_prefetch(Vp + nrow, 0, 3);
    }

    // ---- scores = Q K^T : B column n = key row (contiguous d in Ks) ----
    v8f sacc[2][4];
#pragma unroll
    for (int ti = 0; ti < 2; ++ti)
#pragma unroll
      for (int kt = 0; kt < 4; ++kt) sacc[ti][kt] = zero8f();
#pragma unroll
    for (int c = 0; c < 2; ++c)
#pragma unroll
      for (int kt = 0; kt < 4; ++kt) {
        v16h bf = load_frag_lds(&Ks[kt * 16 * 72] + c * 32, 72, lane);
#pragma unroll
        for (int ti = 0; ti < 2; ++ti)
          sacc[ti][kt] = wmma_f16f32(af[ti][c], bf, sacc[ti][kt]);
      }

    // ---- online softmax (causal mask from C-layout indices) ----
#pragma unroll
    for (int ti = 0; ti < 2; ++ti) {
#pragma unroll
      for (int r = 0; r < 8; ++r) {
        const int qi = qw + ti * 16 + hf * 8 + r;
        float mx = -3.0e38f;
#pragma unroll
        for (int kt = 0; kt < 4; ++kt) {
          int ki = j0 + kt * 16 + nl;
          float s = sacc[ti][kt][r] * 0.125f;  // 1/sqrt(64)
          s = (ki > qi) ? -3.0e38f : s;
          sacc[ti][kt][r] = s;
          mx = fmaxf(mx, s);
        }
        // row lives across a 16-lane half: butterfly within the half
#pragma unroll
        for (int d = 1; d < 16; d <<= 1) mx = fmaxf(mx, __shfl_xor(mx, d, 32));
        float mn   = fmaxf(m_run[ti][r], mx);
        float corr = __expf(m_run[ti][r] - mn);
        float rs = 0.0f;
#pragma unroll
        for (int kt = 0; kt < 4; ++kt) {
          float p = __expf(sacc[ti][kt][r] - mn);
          sacc[ti][kt][r] = p;
          rs += p;
        }
#pragma unroll
        for (int d = 1; d < 16; d <<= 1) rs += __shfl_xor(rs, d, 32);
        l_run[ti][r] = l_run[ti][r] * corr + rs;
        m_run[ti][r] = mn;
#pragma unroll
        for (int j = 0; j < 4; ++j) cacc[ti][j][r] *= corr;
      }
    }

    // ---- P (C layout, f32) -> per-wave LDS [q][key] f16 ----
    _Float16* Pw = &Ps[wave][0];
#pragma unroll
    for (int ti = 0; ti < 2; ++ti)
#pragma unroll
      for (int kt = 0; kt < 4; ++kt)
#pragma unroll
        for (int r = 0; r < 8; ++r)
          Pw[(ti * 16 + hf * 8 + r) * 72 + kt * 16 + nl] =
              (_Float16)sacc[ti][kt][r];

    // ---- ctx += P @ V : B column n = d row of Vt (contiguous keys) ----
#pragma unroll
    for (int c = 0; c < 2; ++c) {
      v16h pf[2];
#pragma unroll
      for (int ti = 0; ti < 2; ++ti)
        pf[ti] = load_frag_lds(Pw + ti * 16 * 72 + c * 32, 72, lane);
#pragma unroll
      for (int j = 0; j < 4; ++j) {
        v16h vf = load_frag_lds(&Vt[j * 16 * 72] + c * 32, 72, lane);
#pragma unroll
        for (int ti = 0; ti < 2; ++ti)
          cacc[ti][j] = wmma_f16f32(pf[ti], vf, cacc[ti][j]);
      }
    }
  }

  // ---- normalize and write merged-head context (f16) ----
#pragma unroll
  for (int ti = 0; ti < 2; ++ti)
#pragma unroll
    for (int j = 0; j < 4; ++j)
#pragma unroll
      for (int r = 0; r < 8; ++r) {
        float o = cacc[ti][j][r] / l_run[ti][r];
        int s = qw + ti * 16 + hf * 8 + r;
        int d = j * 16 + nl;
        O[((size_t)b * S_ + s) * E_ + h * HD_ + d] = (_Float16)o;
      }
}

// ---------------------------------------------------------------------------
// Host launcher
// ---------------------------------------------------------------------------
extern "C" void kernel_launch(void* const* d_in, const int* in_sizes, int n_in,
                              void* d_out, int out_size, void* d_ws,
                              size_t ws_size, hipStream_t stream) {
  (void)in_sizes; (void)n_in; (void)out_size; (void)ws_size;
  const float* q  = (const float*)d_in[0];
  const float* k  = (const float*)d_in[1];
  const float* v  = (const float*)d_in[2];
  // d_in[3] = causal mask: computed analytically in the kernel, not read.
  const float* Wq = (const float*)d_in[4];
  const float* bq = (const float*)d_in[5];
  const float* Wk = (const float*)d_in[6];
  const float* bk = (const float*)d_in[7];
  const float* Wv = (const float*)d_in[8];
  const float* bv = (const float*)d_in[9];
  const float* Wo = (const float*)d_in[10];
  const float* bo = (const float*)d_in[11];
  float* out = (float*)d_out;

  const size_t EE  = (size_t)E_ * E_;       // 1M
  const size_t BSE = (size_t)B_ * S_ * E_;  // 8M
  _Float16* ws   = (_Float16*)d_ws;
  _Float16* Wq16 = ws;
  _Float16* Wk16 = Wq16 + EE;
  _Float16* Wv16 = Wk16 + EE;
  _Float16* Wo16 = Wv16 + EE;
  _Float16* Q16  = Wo16 + EE;
  _Float16* K16  = Q16 + BSE;
  _Float16* V16  = K16 + BSE;
  _Float16* C16  = V16 + BSE;  // total ws: 72 MB

  // 1) weights -> f16
  {
    dim3 blk(256), grd((unsigned)(EE / 4 / 256));
    cvt_f32_f16_kernel<<<grd, blk, 0, stream>>>(Wq, Wq16, (int)(EE / 4));
    cvt_f32_f16_kernel<<<grd, blk, 0, stream>>>(Wk, Wk16, (int)(EE / 4));
    cvt_f32_f16_kernel<<<grd, blk, 0, stream>>>(Wv, Wv16, (int)(EE / 4));
    cvt_f32_f16_kernel<<<grd, blk, 0, stream>>>(Wo, Wo16, (int)(EE / 4));
  }

  // 2) QKV projections: [8192,1024] @ W^T + b -> f16
  dim3 gblk(128);
  dim3 ggrd(B_ * S_ / 128, E_ / 128);  // (64, 8)
  gemm_xwT_kernel<float, true><<<ggrd, gblk, 0, stream>>>(q, Wq16, bq, Q16);
  gemm_xwT_kernel<float, true><<<ggrd, gblk, 0, stream>>>(k, Wk16, bk, K16);
  gemm_xwT_kernel<float, true><<<ggrd, gblk, 0, stream>>>(v, Wv16, bv, V16);

  // 3) causal flash attention per (b, h)
  dim3 agrd(S_ / 128, B_ * H_);  // (16, 64)
  attn_flash_kernel<<<agrd, gblk, 0, stream>>>(Q16, K16, V16, C16);

  // 4) output projection: ctx(f16) @ Wo^T + bo -> f32 d_out
  gemm_xwT_kernel<_Float16, false><<<ggrd, gblk, 0, stream>>>(C16, Wo16, bo, out);
}
